// FlashAttention3_15822659518957
// MI455X (gfx1250) — matrix operand order repaired and checked by reading, not compiled, not run on hardware
//
#include <hip/hip_runtime.h>

typedef _Float16 v16h __attribute__((ext_vector_type(16)));
typedef _Float16 v8h  __attribute__((ext_vector_type(8)));
typedef _Float16 v4h  __attribute__((ext_vector_type(4)));
typedef float    v8f  __attribute__((ext_vector_type(8)));

#define DIM   2048
#define SEQ   2048
#define HD    128
#define NHEAD 16
#define QK_SCALE 0.08838834764831845f   // 128^-0.5 (folded into Q at projection)
#define EPSV  1e-6f

union F16Frag { v16h v; uint4 u[2]; };
union PkHalf  { unsigned int u; _Float16 h[2]; };

static __device__ inline v8f wmma_f16(v16h a, v16h b, v8f c) {
  return __builtin_amdgcn_wmma_f32_16x16x32_f16(false, a, false, b, (short)0, c, false, false);
}

// async global -> LDS copy, 16 bytes per lane (tracked by ASYNCcnt)
static __device__ inline void async_copy16(unsigned lds_off, const void* gptr) {
  asm volatile("global_load_async_to_lds_b128 %0, %1, off"
               :: "v"(lds_off), "v"((unsigned long long)(uintptr_t)gptr)
               : "memory");
}
static __device__ inline void wait_async0() {
  asm volatile("s_wait_asynccnt 0x0" ::: "memory");
}

// ---------------- kernel 1: fp32 -> f16 cast of x ----------------
__global__ void cvt_f32_f16_kernel(const float* __restrict__ in,
                                   _Float16* __restrict__ out, int n4) {
  int i = blockIdx.x * blockDim.x + threadIdx.x;
  if (i < n4) {
    float4 f = ((const float4*)in)[i];
    v4h o = { (_Float16)f.x, (_Float16)f.y, (_Float16)f.z, (_Float16)f.w };
    ((v4h*)out)[i] = o;
  }
}

// ---------------- kernel 2/4: GEMM  C = A(f16,[2048x2048]) * W^T(f32,[N,K]) ----
// FINAL==0: blockIdx.z selects Wq/Wk/Wv; z==0 -> RoPE*scale, z==1 -> RoPE,
// z==2 -> f16 transposed store (Vt[d][s]).  FINAL==1: Wo, fp32 store to d_out.
template<int FINAL>
__global__ void __launch_bounds__(256)
gemm_kernel(const _Float16* __restrict__ A,
            const float* __restrict__ W0, const float* __restrict__ W1,
            const float* __restrict__ W2,
            _Float16* __restrict__ O0, _Float16* __restrict__ O1,
            _Float16* __restrict__ O2, float* __restrict__ OF,
            const float* __restrict__ fcos, const float* __restrict__ fsin)
{
  __shared__ __align__(16) _Float16 ldsW[2][128 * 32];  // double-buffered 2x8 KB
  const int tid  = threadIdx.x;
  const int wave = tid >> 5;
  const int lane = tid & 31;
  const int ll   = lane & 15;
  const int hi   = lane >> 4;
  const int z    = FINAL ? 3 : blockIdx.z;
  const float* W = FINAL ? W0 : (z == 0 ? W0 : (z == 1 ? W1 : W2));
  const int row0  = blockIdx.x * 256 + wave * 32;       // 32 rows per wave
  const int nBase = blockIdx.y * 128;

  const v8f vzero = {0.f,0.f,0.f,0.f,0.f,0.f,0.f,0.f};
  v8f acc[2][8];
  #pragma unroll
  for (int i = 0; i < 2; i++)
    #pragma unroll
    for (int j = 0; j < 8; j++) acc[i][j] = vzero;

  auto stageW = [&](int kb, int buf) {
    #pragma unroll
    for (int t = 0; t < 2; t++) {
      int idx = tid + t * 256;
      int r = idx >> 2, c = idx & 3;
      const float* src = W + (size_t)(nBase + r) * DIM + kb + c * 8;
      float4 f0 = ((const float4*)src)[0];
      float4 f1 = ((const float4*)src)[1];
      v8h hv = { (_Float16)f0.x,(_Float16)f0.y,(_Float16)f0.z,(_Float16)f0.w,
                 (_Float16)f1.x,(_Float16)f1.y,(_Float16)f1.z,(_Float16)f1.w };
      *(v8h*)&ldsW[buf][r * 32 + c * 8] = hv;
    }
  };

  stageW(0, 0);
  __syncthreads();

  for (int i = 0; i < 64; i++) {
    const int kb  = i * 32;
    const int cur = i & 1;
    if (i + 1 < 64) stageW(kb + 32, cur ^ 1);   // prefetch next tile (overlaps WMMA)

    F16Frag a[2];
    #pragma unroll
    for (int mt = 0; mt < 2; mt++) {
      const _Float16* ap = A + (size_t)(row0 + mt * 16 + ll) * DIM + kb + hi * 16;
      a[mt].u[0] = *(const uint4*)ap;          // k chunk [hi*16 .. +8)
      a[mt].u[1] = *(const uint4*)(ap + 8);    // k chunk [hi*16+8 .. +8)
    }
    #pragma unroll
    for (int nt = 0; nt < 8; nt++) {
      F16Frag b;
      const _Float16* bp = &ldsW[cur][(nt * 16 + ll) * 32 + hi * 16];
      b.u[0] = *(const uint4*)bp;
      b.u[1] = *(const uint4*)(bp + 8);
      acc[0][nt] = wmma_f16(a[0].v, b.v, acc[0][nt]);
      acc[1][nt] = wmma_f16(a[1].v, b.v, acc[1][nt]);
    }
    __syncthreads();
  }

  // ---------------- epilogue ----------------
  if (FINAL) {
    #pragma unroll
    for (int mt = 0; mt < 2; mt++) {
      int rowT = row0 + mt * 16 + hi * 8;
      #pragma unroll
      for (int nt = 0; nt < 8; nt++) {
        int n = nBase + nt * 16 + ll;
        #pragma unroll
        for (int r = 0; r < 8; r++)
          OF[(size_t)(rowT + r) * DIM + n] = acc[mt][nt][r];
      }
    }
  } else if (z < 2) {
    // RoPE: pairs (2f,2f+1) sit in adjacent lanes (even/odd column).
    // For Q (z==0) also fold in the attention scale 128^-0.5.
    _Float16* out = (z == 0) ? O0 : O1;
    const float post = (z == 0) ? QK_SCALE : 1.0f;
    #pragma unroll
    for (int mt = 0; mt < 2; mt++) {
      int rowT = row0 + mt * 16 + hi * 8;
      #pragma unroll
      for (int nt = 0; nt < 8; nt++) {
        int n = nBase + nt * 16 + ll;
        int f = (n & 127) >> 1;
        bool even = ((n & 1) == 0);
        #pragma unroll
        for (int r = 0; r < 8; r++) {
          int row = rowT + r;
          float c  = fcos[row * 64 + f];
          float s  = fsin[row * 64 + f];
          float a0 = acc[mt][nt][r];
          float pr = __shfl_xor(a0, 1, 32);
          float val = (even ? (a0 * c - pr * s) : (pr * s + a0 * c)) * post;
          out[(size_t)row * DIM + n] = (_Float16)val;
        }
      }
    }
  } else {
    // V stored transposed: Vt[d][s]; register pairs are adjacent rows -> pack
    #pragma unroll
    for (int mt = 0; mt < 2; mt++) {
      int rowT = row0 + mt * 16 + hi * 8;
      #pragma unroll
      for (int nt = 0; nt < 8; nt++) {
        int n = nBase + nt * 16 + ll;     // d index
        #pragma unroll
        for (int r = 0; r < 8; r += 2) {
          PkHalf pk;
          pk.h[0] = (_Float16)acc[mt][nt][r];
          pk.h[1] = (_Float16)acc[mt][nt][r + 1];
          *(unsigned int*)(O2 + (size_t)n * SEQ + rowT + r) = pk.u;
        }
      }
    }
  }
}

// ---------------- kernel 3: blockwise attention -----------------
// block = 8 waves, 128 query rows, one head; loop over 32 KV blocks of 64.
// K/Vt blocks staged via async global->LDS DMA, double-buffered.
__global__ void __launch_bounds__(256)
attn_kernel(const _Float16* __restrict__ Qh, const _Float16* __restrict__ Kh,
            const _Float16* __restrict__ Vt, _Float16* __restrict__ Oh)
{
  __shared__ __align__(16) _Float16 ldsK[2][64 * 128];   // 2 x 16 KB
  __shared__ __align__(16) _Float16 ldsV[2][128 * 64];   // 2 x 16 KB
  __shared__ __align__(16) _Float16 ldsP[8][16 * 64];    // 16 KB per-wave P scratch
  const int tid = threadIdx.x, wave = tid >> 5, lane = tid & 31;
  const int ll = lane & 15, hi = lane >> 4;
  const int head  = blockIdx.y;
  const int qrow0 = blockIdx.x * 128 + wave * 16;
  const v8f vzero = {0.f,0.f,0.f,0.f,0.f,0.f,0.f,0.f};

  auto issueKV = [&](int jb, int buf) {
    #pragma unroll
    for (int t = 0; t < 4; t++) {
      int idx = tid + t * 256;
      int r = idx >> 4, c = idx & 15;
      async_copy16((unsigned)(uintptr_t)&ldsK[buf][r * 128 + c * 8],
                   Kh + (size_t)(jb * 64 + r) * DIM + head * HD + c * 8);
    }
    #pragma unroll
    for (int t = 0; t < 4; t++) {
      int idx = tid + t * 256;
      int r = idx >> 3, c = idx & 7;
      async_copy16((unsigned)(uintptr_t)&ldsV[buf][r * 64 + c * 8],
                   Vt + (size_t)(head * HD + r) * SEQ + jb * 64 + c * 8);
    }
  };

  // Q strip (16 rows x 128, pre-scaled) in registers for whole kernel
  F16Frag q[4];
  #pragma unroll
  for (int kk = 0; kk < 4; kk++) {
    const _Float16* qp = Qh + (size_t)(qrow0 + ll) * DIM + head * HD + kk * 32 + hi * 16;
    q[kk].u[0] = *(const uint4*)qp;
    q[kk].u[1] = *(const uint4*)(qp + 8);
  }
  v8f o[8];
  #pragma unroll
  for (int i = 0; i < 8; i++) o[i] = vzero;
  float Lsum[8];
  #pragma unroll
  for (int r = 0; r < 8; r++) Lsum[r] = 0.f;

  issueKV(0, 0);                      // prologue DMA

  for (int jb = 0; jb < 32; jb++) {
    const int cur = jb & 1;
    wait_async0();                    // my DMA for buf[cur] done
    __syncthreads();                  // everyone's DMA done; prev compute done
    if (jb + 1 < 32) issueKV(jb + 1, cur ^ 1);   // overlap next block's DMA

    // S = Q K^T  (16 x 64, over hd=128); scale pre-folded into Q
    v8f sa[4];
    #pragma unroll
    for (int nt = 0; nt < 4; nt++) {
      sa[nt] = vzero;
      #pragma unroll
      for (int kk = 0; kk < 4; kk++) {
        F16Frag b;
        const _Float16* bp = &ldsK[cur][(nt * 16 + ll) * 128 + kk * 32 + hi * 16];
        b.u[0] = *(const uint4*)bp;
        b.u[1] = *(const uint4*)(bp + 8);
        sa[nt] = wmma_f16(q[kk].v, b.v, sa[nt]);
      }
    }

    // per-KV-block row max (reference semantics: no cross-block rescale)
    float rmax[8];
    #pragma unroll
    for (int r = 0; r < 8; r++) {
      float m = fmaxf(fmaxf(sa[0][r], sa[1][r]), fmaxf(sa[2][r], sa[3][r]));
      #pragma unroll
      for (int msk = 1; msk < 16; msk <<= 1)
        m = fmaxf(m, __shfl_xor(m, msk, 32));
      rmax[r] = m;
    }
    #pragma unroll
    for (int nt = 0; nt < 4; nt++)
      #pragma unroll
      for (int r = 0; r < 8; r++) {
        float p = __expf(sa[nt][r] - rmax[r]);
        sa[nt][r] = p;
        Lsum[r] += p;
      }

    // C-layout P -> LDS -> A-layout fragments (per-wave private scratch)
    _Float16* myP = &ldsP[wave][0];
    #pragma unroll
    for (int nt = 0; nt < 4; nt++)
      #pragma unroll
      for (int r = 0; r < 8; r++)
        myP[(r + hi * 8) * 64 + nt * 16 + ll] = (_Float16)sa[nt][r];
    asm volatile("s_wait_dscnt 0x0" ::: "memory");
    F16Frag pa[2];
    #pragma unroll
    for (int kk = 0; kk < 2; kk++) {
      const _Float16* pp = myP + ll * 64 + kk * 32 + hi * 16;
      pa[kk].u[0] = *(const uint4*)pp;
      pa[kk].u[1] = *(const uint4*)(pp + 8);
    }

    // O += P * V   (B fragments contiguous thanks to V^T layout)
    #pragma unroll
    for (int nt = 0; nt < 8; nt++)
      #pragma unroll
      for (int kk = 0; kk < 2; kk++) {
        F16Frag b;
        const _Float16* bp = &ldsV[cur][(nt * 16 + ll) * 64 + kk * 32 + hi * 16];
        b.u[0] = *(const uint4*)bp;
        b.u[1] = *(const uint4*)(bp + 8);
        o[nt] = wmma_f16(pa[kk].v, b.v, o[nt]);
      }
  }

  // normalize by (sum L + eps) and store O as f16 row-major
  #pragma unroll
  for (int r = 0; r < 8; r++) {
    float s = Lsum[r];
    #pragma unroll
    for (int msk = 1; msk < 16; msk <<= 1) s += __shfl_xor(s, msk, 32);
    Lsum[r] = 1.0f / (s + EPSV);
  }
  #pragma unroll
  for (int nt = 0; nt < 8; nt++) {
    int col = head * HD + nt * 16 + ll;
    #pragma unroll
    for (int r = 0; r < 8; r++) {
      float val = o[nt][r] * Lsum[r];
      Oh[(size_t)(qrow0 + hi * 8 + r) * DIM + col] = (_Float16)val;
    }
  }
}

extern "C" void kernel_launch(void* const* d_in, const int* in_sizes, int n_in,
                              void* d_out, int out_size, void* d_ws, size_t ws_size,
                              hipStream_t stream) {
  const float* x    = (const float*)d_in[0];
  const float* fcos = (const float*)d_in[1];
  const float* fsin = (const float*)d_in[2];
  const float* Wq   = (const float*)d_in[3];
  const float* Wk   = (const float*)d_in[4];
  const float* Wv   = (const float*)d_in[5];
  const float* Wo   = (const float*)d_in[6];
  float* out = (float*)d_out;

  const size_t MAT = (size_t)DIM * SEQ;          // 4M elements
  _Float16* xh = (_Float16*)d_ws;                // x   f16        8 MB
  _Float16* Qh = xh + MAT;                       // Q   f16 (rope*scale)
  _Float16* Kh = Qh + MAT;                       // K   f16 (rope)
  _Float16* Vt = Kh + MAT;                       // V^T f16
  _Float16* Oh = Vt + MAT;                       // O   f16

  cvt_f32_f16_kernel<<<4096, 256, 0, stream>>>(x, xh, (int)(MAT / 4));
  gemm_kernel<0><<<dim3(8, 16, 3), 256, 0, stream>>>(
      xh, Wq, Wk, Wv, Qh, Kh, Vt, nullptr, fcos, fsin);
  attn_kernel<<<dim3(16, 16), 256, 0, stream>>>(Qh, Kh, Vt, Oh);
  gemm_kernel<1><<<dim3(8, 16, 1), 256, 0, stream>>>(
      Oh, Wo, Wo, Wo, nullptr, nullptr, nullptr, out, fcos, fsin);
}
